// Net_73203422593380
// MI455X (gfx1250) — matrix operand order repaired and checked by reading, not compiled, not run on hardware
//
#include <hip/hip_runtime.h>
#include <hip/hip_bf16.h>
#include <math.h>

// Problem constants (from reference)
#define BG 256       // groups
#define XD 784       // input dim (K of GEMM1)
#define ZD 32        // hidden dim
#define NC 10        // classes
#define LL 512       // sequence length (N)
#define KP 800       // K padded to multiple of 32
#define KT 32        // K chunk per WMMA step
#define NCHUNK (KP / KT)  // 25
#define APAD 808     // W1 LDS row stride in bf16 (1616B: 16B-aligned, conflict-free)
#define XSTR 36      // x chunk LDS row stride in f32 (144B: 16B-aligned, conflict-free)
#define HSTR 40      // h LDS row stride in bf16 (80B: 16B-aligned, conflict-free)

typedef __bf16 v16bf __attribute__((ext_vector_type(16)));
typedef __bf16 v8bf  __attribute__((ext_vector_type(8)));
typedef float  v8f   __attribute__((ext_vector_type(8)));
typedef float  v4f   __attribute__((ext_vector_type(4)));

union FragU { v16bf v; v8bf h[2]; };

// ---- dynamic LDS carve-out (bytes) ----
#define OFF_A    0                            // W1[g] bf16 [32][808]    = 51712
#define OFF_X0   (OFF_A  + ZD*APAD*2)         // x chunk f32 [512][36]   = 73728
#define OFF_X1   (OFF_X0 + LL*XSTR*4)         // x chunk f32 (buffer 1)  = 73728
#define OFF_HT   (OFF_X1 + LL*XSTR*4)         // h bf16 [512][40]        = 40960
#define OFF_W2   (OFF_HT + LL*HSTR*2)         // W2 padded bf16 [16][32] = 1024
#define OFF_O    (OFF_W2 + 16*32*2)           // o f32 [10][512]         = 20480
#define OFF_B1   (OFF_O  + NC*LL*4)           // b1 f32 [32]
#define OFF_B2   (OFF_B1 + ZD*4)              // b2 f32 [16]
#define OFF_RED  (OFF_B2 + 16*4)              // reduction f32 [512]
#define SMEM_BYTES (OFF_RED + 512*4)          // ~263.9 KB (< 320 KB/WGP)

// A fragment (16x32 bf16, MxK): lane<16 holds K 0..15, lane>=16 holds
// K 16..31 of row M = lane&15 (matches B fragment k order).
__device__ inline v16bf lds_frag_a(const __bf16* rowp, int lane) {
    FragU f;
    const __bf16* p = rowp + ((lane >= 16) ? 16 : 0);
    f.h[0] = *(const v8bf*)(p);
    f.h[1] = *(const v8bf*)(p + 8);
    return f.v;
}
// B fragment (32x16 bf16, KxN) from transposed bf16 LDS [n][k]:
// lane<16 holds K 0..15, lane>=16 holds K 16..31 of column N = lane&15.
__device__ inline v16bf lds_frag_b_bf16(const __bf16* rowp, int lane) {
    FragU f;
    const __bf16* p = rowp + ((lane >= 16) ? 16 : 0);
    f.h[0] = *(const v8bf*)(p);
    f.h[1] = *(const v8bf*)(p + 8);
    return f.v;
}
// B fragment from transposed f32 LDS [n][k] (async-staged x), converting to bf16.
__device__ inline v16bf lds_frag_b_f32(const float* rowp, int lane) {
    const float* p = rowp + ((lane >= 16) ? 16 : 0);
    v4f c0 = *(const v4f*)(p);
    v4f c1 = *(const v4f*)(p + 4);
    v4f c2 = *(const v4f*)(p + 8);
    v4f c3 = *(const v4f*)(p + 12);
    v16bf r;
    #pragma unroll
    for (int i = 0; i < 4; ++i) {
        r[i]      = (__bf16)c0[i];
        r[4 + i]  = (__bf16)c1[i];
        r[8 + i]  = (__bf16)c2[i];
        r[12 + i] = (__bf16)c3[i];
    }
    return r;
}

// Async global->LDS copy, 4 bytes per lane (GV mode: per-lane LDS dest VGPR,
// 64-bit global address VGPR pair). Tracked by ASYNCcnt.
__device__ inline void async_g2l_b32(const float* gptr, float* lptr) {
    unsigned lds = (unsigned)(unsigned long long)lptr;  // LDS aperture: low 32 bits
    asm volatile("global_load_async_to_lds_b32 %0, %1, off"
                 :: "v"(lds), "v"(gptr) : "memory");
}
__device__ inline void wait_asynccnt0() {
    asm volatile("s_wait_asynccnt 0x0" ::: "memory");
}

// Stage one K-chunk of x[g], transposed [n][k], via async copies.
// Thread t owns column n=t; 32 async b32 per thread, each instruction moving a
// coalesced 128B global line. Tail rows (k>=784) are zero-filled via LDS stores.
__device__ inline void stage_chunk_async(const float* xg, int kc, float* buf, int t) {
    float* row = buf + t * XSTR;
    #pragma unroll
    for (int k = 0; k < KT; ++k) {
        int kg = kc + k;
        if (kg < XD) {
            async_g2l_b32(xg + (size_t)kg * LL + t, row + k);
        } else {
            row[k] = 0.0f;   // uniform branch: whole chunk-row invalid
        }
    }
}

__global__ __launch_bounds__(512, 1)
void hypermlp_fused(const float* __restrict__ x,   const float* __restrict__ W1,
                    const float* __restrict__ b1,  const float* __restrict__ beta,
                    const float* __restrict__ W2,  const float* __restrict__ b2,
                    float* __restrict__ out)
{
    extern __shared__ char smem[];
    __bf16* Abuf = (__bf16*)(smem + OFF_A);
    float*  xT0  = (float* )(smem + OFF_X0);
    float*  xT1  = (float* )(smem + OFF_X1);
    __bf16* hT   = (__bf16*)(smem + OFF_HT);
    __bf16* w2a  = (__bf16*)(smem + OFF_W2);
    float*  oBuf = (float* )(smem + OFF_O);
    float*  b1s  = (float* )(smem + OFF_B1);
    float*  b2s  = (float* )(smem + OFF_B2);
    float*  red  = (float* )(smem + OFF_RED);

    const int g    = blockIdx.x;
    const int t    = threadIdx.x;
    const int lane = t & 31;            // wave32
    const int wv   = t >> 5;            // 16 waves
    const int ln15 = lane & 15;
    const int hi8  = (lane >= 16) ? 8 : 0;

    const float* xg = x + (size_t)g * XD * LL;

    // Kick chunk 0's async fill first so it overlaps weight staging.
    stage_chunk_async(xg, 0, xT0, t);

    // ---- stage per-group weights/biases into LDS ----
    const float* W1g = W1 + (size_t)g * ZD * XD;
    // zero the K padding region (k = 784..807) of each of the 32 rows
    for (int idx = t; idx < ZD * (APAD - XD); idx += 512) {
        int z = idx / (APAD - XD), k = XD + idx % (APAD - XD);
        Abuf[z * APAD + k] = (__bf16)0.0f;
    }
    // bulk of W1 via float4 global loads -> 4 packed bf16 stores
    for (int idx = t; idx < ZD * (XD / 4); idx += 512) {        // 6272 float4s
        int z = idx / (XD / 4), k4 = idx % (XD / 4);
        v4f w = *(const v4f*)(W1g + z * XD + 4 * k4);
        __bf16* d = Abuf + z * APAD + 4 * k4;
        d[0] = (__bf16)w[0]; d[1] = (__bf16)w[1];
        d[2] = (__bf16)w[2]; d[3] = (__bf16)w[3];
    }
    {   // W2 padded 10->16 rows: exactly 512 elements, one per thread
        int c = t >> 5, z = t & 31;
        float w = (c < NC) ? W2[((size_t)g * NC + c) * ZD + z] : 0.0f;
        w2a[c * 32 + z] = (__bf16)w;
    }
    if (t < ZD) b1s[t] = b1[g * ZD + t];
    if (t < 16) b2s[t] = (t < NC) ? b2[g * NC + t] : 0.0f;
    const float sp = log1pf(__expf(beta[g]));   // softplus(beta)

    wait_asynccnt0();
    __syncthreads();

    // ---- GEMM1: h = W1[g] @ x[g]; double-buffered async K-stream ----
    v8f acc00 = {}, acc01 = {}, acc10 = {}, acc11 = {};
    const int nb = 32 * wv;                     // wave's 2 N-tiles
    const __bf16* aRow0 = Abuf + ln15 * APAD;
    const __bf16* aRow1 = Abuf + (16 + ln15) * APAD;

    #pragma unroll 1
    for (int c = 0; c < NCHUNK; ++c) {
        float* cur = (c & 1) ? xT1 : xT0;
        float* nxt = (c & 1) ? xT0 : xT1;
        // issue next chunk's async copies under this chunk's WMMAs
        if (c + 1 < NCHUNK) stage_chunk_async(xg, (c + 1) * KT, nxt, t);

        int kc = c * KT;
        v16bf a0  = lds_frag_a(aRow0 + kc, lane);
        v16bf a1  = lds_frag_a(aRow1 + kc, lane);
        v16bf bb0 = lds_frag_b_f32(cur + (nb + ln15)      * XSTR, lane);
        v16bf bb1 = lds_frag_b_f32(cur + (nb + 16 + ln15) * XSTR, lane);

        acc00 = __builtin_amdgcn_wmma_f32_16x16x32_bf16(false, a0, false, bb0, (short)0, acc00, false, false);
        acc10 = __builtin_amdgcn_wmma_f32_16x16x32_bf16(false, a1, false, bb0, (short)0, acc10, false, false);
        acc01 = __builtin_amdgcn_wmma_f32_16x16x32_bf16(false, a0, false, bb1, (short)0, acc01, false, false);
        acc11 = __builtin_amdgcn_wmma_f32_16x16x32_bf16(false, a1, false, bb1, (short)0, acc11, false, false);

        wait_asynccnt0();     // our fills of `nxt` are done
        __syncthreads();      // everyone done reading `cur` / filling `nxt`
    }

    // ---- bias + GroupSwish, stage h transposed (bf16) for GEMM2 ----
    {
        const int n0 = nb + ln15, n1 = nb + 16 + ln15;
        #pragma unroll
        for (int v = 0; v < 8; ++v) {
            int m0 = v + hi8;        // rows 0..15 (M-tile 0)
            int m1 = 16 + v + hi8;   // rows 16..31 (M-tile 1)
            float h;
            h = acc00[v] + b1s[m0];
            hT[n0 * HSTR + m0] = (__bf16)(h * (1.0f / (1.0f + __expf(-h * sp))) * (1.0f / 1.1f));
            h = acc10[v] + b1s[m1];
            hT[n0 * HSTR + m1] = (__bf16)(h * (1.0f / (1.0f + __expf(-h * sp))) * (1.0f / 1.1f));
            h = acc01[v] + b1s[m0];
            hT[n1 * HSTR + m0] = (__bf16)(h * (1.0f / (1.0f + __expf(-h * sp))) * (1.0f / 1.1f));
            h = acc11[v] + b1s[m1];
            hT[n1 * HSTR + m1] = (__bf16)(h * (1.0f / (1.0f + __expf(-h * sp))) * (1.0f / 1.1f));
        }
    }
    __syncthreads();

    // ---- GEMM2: o = W2pad[16x32] @ h[32x512], one WMMA per 16-wide N tile ----
    {
        v16bf aw = lds_frag_a(w2a + ln15 * 32, lane);
        #pragma unroll
        for (int j = 0; j < 2; ++j) {
            int ntile = 2 * wv + j;
            v16bf bh = lds_frag_b_bf16(hT + (ntile * 16 + ln15) * HSTR, lane);
            v8f o = {};
            o = __builtin_amdgcn_wmma_f32_16x16x32_bf16(false, aw, false, bh, (short)0, o, false, false);
            int n = ntile * 16 + ln15;
            #pragma unroll
            for (int v = 0; v < 8; ++v) {
                int cc = v + hi8;
                if (cc < NC) oBuf[cc * LL + n] = o[v] + b2s[cc];
            }
        }
    }
    __syncthreads();

    // ---- block-local softmax over the group's 10*512 = 5120 outputs ----
    float mx = -3.4e38f;
    for (int i = t; i < NC * LL; i += 512) mx = fmaxf(mx, oBuf[i]);
    red[t] = mx; __syncthreads();
    for (int s = 256; s > 0; s >>= 1) { if (t < s) red[t] = fmaxf(red[t], red[t + s]); __syncthreads(); }
    const float gmax = red[0]; __syncthreads();

    float sm = 0.0f;
    for (int i = t; i < NC * LL; i += 512) sm += __expf(oBuf[i] - gmax);
    red[t] = sm; __syncthreads();
    for (int s = 256; s > 0; s >>= 1) { if (t < s) red[t] += red[t + s]; __syncthreads(); }
    const float inv = 1.0f / red[0]; __syncthreads();

    float* og = out + (size_t)g * NC * LL;
    for (int i = t; i < NC * LL; i += 512) og[i] = __expf(oBuf[i] - gmax) * inv;
}

extern "C" void kernel_launch(void* const* d_in, const int* in_sizes, int n_in,
                              void* d_out, int out_size, void* d_ws, size_t ws_size,
                              hipStream_t stream) {
    (void)in_sizes; (void)n_in; (void)out_size; (void)d_ws; (void)ws_size;
    const float* x    = (const float*)d_in[0];
    const float* W1   = (const float*)d_in[1];
    const float* b1   = (const float*)d_in[2];
    const float* beta = (const float*)d_in[3];
    const float* W2   = (const float*)d_in[4];
    const float* b2   = (const float*)d_in[5];
    float* out = (float*)d_out;

    hypermlp_fused<<<dim3(BG), dim3(512), SMEM_BYTES, stream>>>(x, W1, b1, beta, W2, b2, out);
}
